// CondLaneHead_72009421684834
// MI455X (gfx1250) — compile-verified, matrix operand order checked by reading, not run on hardware
//
#include <hip/hip_runtime.h>
#include <hip/hip_bf16.h>

typedef __attribute__((ext_vector_type(16))) _Float16 v16h;
typedef __attribute__((ext_vector_type(8)))  _Float16 v8h;
typedef __attribute__((ext_vector_type(8)))  float    v8f;

// ---------------------------------------------------------------------------
// Problem constants (from reference setup_inputs)
// ---------------------------------------------------------------------------
#define NB     32      // batch
#define CIN_A  256     // heatmap conv input channels
#define H_A    20
#define W_A    50
#define HW_A   (H_A*W_A)          // 1000
#define CH     64      // hidden / output channels of both convs
#define CIN_B  128     // mask conv input channels
#define H_B    40
#define W_B    100
#define HW_B   (H_B*W_B)          // 4000
#define NINST  128
#define NPAR   134                // 2*(64+2+1)

// ---------------------------------------------------------------------------
// Activation pre-transform: NCHW f32 -> "NHWC" f16 (channel innermost), with
// one extra zero-filled position row at p == HW per batch. All halo /
// out-of-tile accesses in the conv kernels redirect to that zero row, so the
// hot loop issues only unconditional, in-bounds vector loads.
// ---------------------------------------------------------------------------
template <int CIN, int HW>
__global__ __launch_bounds__(256) void nhwc_f16(const float* __restrict__ x,
                                                _Float16* __restrict__ xt) {
    int tid = blockIdx.x * blockDim.x + threadIdx.x;
    int total = NB * (HW + 1);
    if (tid >= total) return;
    int b = tid / (HW + 1);
    int p = tid - b * (HW + 1);
    _Float16* dst = xt + (size_t)tid * CIN;
    if (p == HW) {                       // zero pad row
        v8h z = {};
#pragma unroll
        for (int c = 0; c < CIN; c += 8) *(v8h*)(dst + c) = z;
        return;
    }
    const float* src = x + (size_t)b * CIN * HW + p;
    for (int c = 0; c < CIN; c += 8) {
        v8h v;
#pragma unroll
        for (int j = 0; j < 8; ++j) v[j] = (_Float16)src[(size_t)(c + j) * HW];
        *(v8h*)(dst + c) = v;
    }
}

// ---------------------------------------------------------------------------
// Weight pre-pack: (Cout=64, Cin, 3, 3) fp32  ->  f16 WMMA B-fragments.
// Fragment layout index: ((tap*KSTEPS + ks)*4 + nt)*32 + lane, 16 halfs each.
// Per-lane B mapping (mirrors documented 16-bit A layout):
//   n = nt*16 + (lane&15),  k = ks*32 + (i<8 ? i : i+8) + (lane>=16 ? 8 : 0)
// ---------------------------------------------------------------------------
__global__ __launch_bounds__(256) void pack_wgts(const float* __restrict__ w,
                                                 _Float16* __restrict__ out,
                                                 int Cin) {
    int ksteps = Cin >> 5;
    int total  = 9 * ksteps * 4 * 32;
    int tid = blockIdx.x * blockDim.x + threadIdx.x;
    if (tid >= total) return;
    int lane = tid & 31;
    int nt   = (tid >> 5) & 3;
    int rest = tid >> 7;
    int ks   = rest % ksteps;
    int tap  = rest / ksteps;
    int n    = nt * 16 + (lane & 15);
    int khi  = (lane >= 16) ? 8 : 0;
    v16h v;
#pragma unroll
    for (int i = 0; i < 16; ++i) {
        int k = ks * 32 + (i < 8 ? i : i + 8) + khi;
        v[i] = (_Float16)w[((size_t)(n * Cin + k)) * 9 + tap];
    }
    ((v16h*)out)[tid] = v;
}

// ---------------------------------------------------------------------------
// Core implicit-GEMM: one wave -> 32 output positions x 64 channels
// (2 M-tiles x 4 N-tiles = 8 f32 accumulators).
// Tap-decomposed 3x3 conv flattened to a single (tap,ks) step loop with
// one-step-ahead register prefetch: step s+1's 12 b128 loads are issued
// before step s's 8 WMMAs, so WMMA operands are a full iteration old and
// waits need not drain the load queue.
// A-fragment per-lane mapping (ISA 16-bit A 16x32 layout):
//   M = lane&15, k(i) = (i<8 ? i : i+8) + (lane>=16 ? 8 : 0)
// ---------------------------------------------------------------------------
template <int CIN, int H, int W>
struct ConvFrags {
    v8h  a00, a01, a10, a11;   // two A fragments (low/high 8-half groups)
    v16h b0, b1, b2, b3;       // four B fragments
};

template <int CIN, int H, int W>
__device__ inline void conv_accum2(const _Float16* __restrict__ xt_b, // batch base, (HW+1)*CIN halfs
                                   int ptile, v8f acc[8],
                                   const _Float16* __restrict__ packed) {
    const int HW     = H * W;
    const int KSTEPS = CIN >> 5;
    const int STEPS  = 9 * KSTEPS;
    int lane = threadIdx.x & 31;
    int m    = lane & 15;
    int khi  = (lane >= 16) ? 8 : 0;
    int p0 = ptile + m,      p1 = ptile + 16 + m;
    int y0 = p0 / W, x0 = p0 - y0 * W;
    int y1 = p1 / W, x1 = p1 - y1 * W;
    bool in0 = p0 < HW, in1 = p1 < HW;

    // per-(tap, position) halo offset; invalid -> zero pad row at p == HW
    auto aoff = [&](int tap, int y, int x, bool pin) -> size_t {
        int dy = tap / 3 - 1;
        int dx = tap - (tap / 3) * 3 - 1;
        int ys = y + dy, xs = x + dx;
        bool valid = pin && ((unsigned)ys < (unsigned)H) && ((unsigned)xs < (unsigned)W);
        int poff = valid ? (ys * W + xs) : HW;
        return (size_t)poff * CIN + khi;
    };

    auto load_step = [&](int step) -> ConvFrags<CIN, H, W> {
        int tap = step / KSTEPS;            // KSTEPS is a power of two
        int ks  = step - tap * KSTEPS;
        ConvFrags<CIN, H, W> f;
        const _Float16* r0 = xt_b + aoff(tap, y0, x0, in0) + ks * 32;
        const _Float16* r1 = xt_b + aoff(tap, y1, x1, in1) + ks * 32;
        f.a00 = *(const v8h*)(r0);          // k = ks*32+khi+0..7
        f.a01 = *(const v8h*)(r0 + 16);     // k = ks*32+khi+16..23
        f.a10 = *(const v8h*)(r1);
        f.a11 = *(const v8h*)(r1 + 16);
        const v16h* bf = (const v16h*)packed + ((size_t)(tap * KSTEPS + ks) * 4) * 32 + lane;
        f.b0 = bf[0];  f.b1 = bf[32];  f.b2 = bf[64];  f.b3 = bf[96];
        return f;
    };

    auto do_wmma = [&](const ConvFrags<CIN, H, W>& f) {
        v16h af0 = __builtin_shufflevector(f.a00, f.a01, 0, 1, 2, 3, 4, 5, 6, 7,
                                           8, 9, 10, 11, 12, 13, 14, 15);
        v16h af1 = __builtin_shufflevector(f.a10, f.a11, 0, 1, 2, 3, 4, 5, 6, 7,
                                           8, 9, 10, 11, 12, 13, 14, 15);
        acc[0] = __builtin_amdgcn_wmma_f32_16x16x32_f16(false, af0, false, f.b0, (short)0, acc[0], false, false);
        acc[1] = __builtin_amdgcn_wmma_f32_16x16x32_f16(false, af0, false, f.b1, (short)0, acc[1], false, false);
        acc[2] = __builtin_amdgcn_wmma_f32_16x16x32_f16(false, af0, false, f.b2, (short)0, acc[2], false, false);
        acc[3] = __builtin_amdgcn_wmma_f32_16x16x32_f16(false, af0, false, f.b3, (short)0, acc[3], false, false);
        acc[4] = __builtin_amdgcn_wmma_f32_16x16x32_f16(false, af1, false, f.b0, (short)0, acc[4], false, false);
        acc[5] = __builtin_amdgcn_wmma_f32_16x16x32_f16(false, af1, false, f.b1, (short)0, acc[5], false, false);
        acc[6] = __builtin_amdgcn_wmma_f32_16x16x32_f16(false, af1, false, f.b2, (short)0, acc[6], false, false);
        acc[7] = __builtin_amdgcn_wmma_f32_16x16x32_f16(false, af1, false, f.b3, (short)0, acc[7], false, false);
    };

    ConvFrags<CIN, H, W> cur = load_step(0);
#pragma unroll 2
    for (int step = 0; step < STEPS - 1; ++step) {
        ConvFrags<CIN, H, W> nxt = load_step(step + 1);  // prefetch next step
        do_wmma(cur);                                    // consume previous loads
        cur = nxt;
    }
    do_wmma(cur);                                        // epilogue
}

// ---------------------------------------------------------------------------
// Heatmap branch, fully fused: conv1(3x3,256->64)+bias+ReLU, then conv2(1x1,
// 64->1)+bias. The 1x1 conv is an N-reduction over the WMMA accumulators:
// per-lane dot with w2, then xor-butterfly over each 16-lane half.
// C layout: value(M = r + 8*(lane>=16), N = nt*16 + lane&15) in acc[t*4+nt][r]
// for position tile t (positions ptile + t*16 + M).
// ---------------------------------------------------------------------------
__global__ __launch_bounds__(256) void conv_hm_fused(
    const _Float16* __restrict__ xtA, const _Float16* __restrict__ packed,
    const float* __restrict__ b1, const float* __restrict__ w2,
    const float* __restrict__ b2, float* __restrict__ hm_out) {
    int wave  = threadIdx.x >> 5;
    int lane  = threadIdx.x & 31;
    int b     = blockIdx.x >> 2;     // 4 tile-blocks per batch (4*256 >= 1000)
    int tb    = blockIdx.x & 3;
    int ptile = tb * 256 + wave * 32;

    const _Float16* xb = xtA + (size_t)b * (HW_A + 1) * CIN_A;
    v8f acc[8] = {};
    conv_accum2<CIN_A, H_A, W_A>(xb, ptile, acc, packed);

    int nlo = lane & 15;
    float s[2][8];
#pragma unroll
    for (int t = 0; t < 2; ++t)
#pragma unroll
        for (int r = 0; r < 8; ++r) s[t][r] = 0.0f;
#pragma unroll
    for (int nt = 0; nt < 4; ++nt) {
        float bias = b1[nt * 16 + nlo];
        float wv   = w2[nt * 16 + nlo];
#pragma unroll
        for (int t = 0; t < 2; ++t)
#pragma unroll
            for (int r = 0; r < 8; ++r) {
                float h = acc[t * 4 + nt][r] + bias;
                h = h > 0.0f ? h : 0.0f;       // ReLU on hidden activations
                s[t][r] += h * wv;             // fused 1x1 conv
            }
    }
#pragma unroll
    for (int msk = 1; msk < 16; msk <<= 1)
#pragma unroll
        for (int t = 0; t < 2; ++t)
#pragma unroll
            for (int r = 0; r < 8; ++r) s[t][r] += __shfl_xor(s[t][r], msk, 32);

    if (nlo == 0) {                        // lanes 0 and 16 write M=0..7 / 8..15
        float bb  = b2[0];
        int mbase = (lane >= 16) ? 8 : 0;
#pragma unroll
        for (int t = 0; t < 2; ++t)
#pragma unroll
            for (int r = 0; r < 8; ++r) {
                int p = ptile + t * 16 + mbase + r;
                if (p < HW_A) hm_out[(size_t)b * HW_A + p] = s[t][r] + bb;
            }
    }
}

// ---------------------------------------------------------------------------
// Mask-feature conv: 3x3, 128->64, +bias, ReLU -> mf[b][c][p] (f16, ws).
// Lane's 8 accumulator rows are 8 consecutive positions at fixed channel ->
// one 16B vector store when fully in-bounds.
// ---------------------------------------------------------------------------
__global__ __launch_bounds__(256) void conv_mask(
    const _Float16* __restrict__ xtB, const _Float16* __restrict__ packed,
    const float* __restrict__ bC, _Float16* __restrict__ mf) {
    int wave  = threadIdx.x >> 5;
    int lane  = threadIdx.x & 31;
    int b     = blockIdx.x >> 4;     // 16 tile-blocks per batch (16*256 >= 4000)
    int tb    = blockIdx.x & 15;
    int ptile = tb * 256 + wave * 32;

    const _Float16* xb = xtB + (size_t)b * (HW_B + 1) * CIN_B;
    v8f acc[8] = {};
    conv_accum2<CIN_B, H_B, W_B>(xb, ptile, acc, packed);

    int nlo   = lane & 15;
    int mbase = (lane >= 16) ? 8 : 0;
#pragma unroll
    for (int nt = 0; nt < 4; ++nt) {
        int n = nt * 16 + nlo;
        float bias = bC[n];
        _Float16* dst = mf + ((size_t)b * CH + n) * HW_B;
#pragma unroll
        for (int t = 0; t < 2; ++t) {
            int pbase = ptile + t * 16 + mbase;
            v8h pk;
#pragma unroll
            for (int r = 0; r < 8; ++r) {
                float v = acc[t * 4 + nt][r] + bias;
                v = v > 0.0f ? v : 0.0f;
                pk[r] = (_Float16)v;
            }
            if (pbase + 8 <= HW_B) {
                *(v8h*)(dst + pbase) = pk;     // 16B coalesced store
            } else {
#pragma unroll
                for (int r = 0; r < 8; ++r)
                    if (pbase + r < HW_B) dst[pbase + r] = pk[r];
            }
        }
    }
}

// ---------------------------------------------------------------------------
// Dynamic per-instance heads: masks/regs = <x_cat, w> + b. Tiny FLOPs,
// bandwidth bound; params staged in LDS, mf reads coalesced across the wave.
// ---------------------------------------------------------------------------
__global__ __launch_bounds__(256) void dyn_heads(
    const _Float16* __restrict__ mf, const float* __restrict__ dynp,
    const int* __restrict__ inst_idx, float* __restrict__ masks,
    float* __restrict__ regs) {
    __shared__ float sw[NPAR];
    int n = blockIdx.y;
    for (int i = threadIdx.x; i < NPAR; i += blockDim.x) sw[i] = dynp[n * NPAR + i];
    __syncthreads();

    int p = blockIdx.x * blockDim.x + threadIdx.x;
    if (p >= HW_B) return;
    int b = inst_idx[n];
    float cx = (float)(p % W_B);
    float cy = (float)(p / W_B);
    const _Float16* src = mf + (size_t)b * CH * HW_B + p;
    float am = sw[0] * cx + sw[1] * cy;       // coord channels of w_mask
    float ar = sw[67] * cx + sw[68] * cy;     // coord channels of w_reg
#pragma unroll 8
    for (int c = 0; c < CH; ++c) {
        float v = (float)src[(size_t)c * HW_B];
        am += sw[2 + c] * v;
        ar += sw[69 + c] * v;
    }
    am += sw[66] - 2.19f;                     // b_mask = dyn[66] - 2.19
    ar += sw[133];                            // b_reg
    masks[(size_t)n * HW_B + p] = am;
    regs [(size_t)n * HW_B + p] = ar;
}

// ---------------------------------------------------------------------------
// Launch
// ---------------------------------------------------------------------------
extern "C" void kernel_launch(void* const* d_in, const int* in_sizes, int n_in,
                              void* d_out, int out_size, void* d_ws, size_t ws_size,
                              hipStream_t stream) {
    const float* f_hm    = (const float*)d_in[0];
    const float* f_mask  = (const float*)d_in[1];
    const float* dynp    = (const float*)d_in[2];
    const int*   inst    = (const int*)  d_in[3];
    const float* hm_w1   = (const float*)d_in[4];
    const float* hm_b1   = (const float*)d_in[5];
    const float* hm_w2   = (const float*)d_in[6];
    const float* hm_b2   = (const float*)d_in[7];
    const float* mask_w  = (const float*)d_in[8];
    const float* mask_b  = (const float*)d_in[9];

    float* out    = (float*)d_out;
    float* hm_out = out;                         // (32,1,20,50)  = 32000
    float* masks  = out + 32000;                 // (128,1,40,100)= 512000
    float* regs   = out + 32000 + 512000;        // (128,1,40,100)= 512000

    // workspace layout (bytes)
    char* ws = (char*)d_ws;
    const size_t packA_off = 0;
    const size_t packA_sz  = (size_t)9 * 8 * 4 * 32 * 16 * sizeof(_Float16); //   294,912
    const size_t packB_off = packA_off + packA_sz;
    const size_t packB_sz  = (size_t)9 * 4 * 4 * 32 * 16 * sizeof(_Float16); //   147,456
    const size_t xtA_off   = packB_off + packB_sz;
    const size_t xtA_sz    = (size_t)NB * (HW_A + 1) * CIN_A * sizeof(_Float16); // 16.4 MB
    const size_t xtB_off   = xtA_off + xtA_sz;
    const size_t xtB_sz    = (size_t)NB * (HW_B + 1) * CIN_B * sizeof(_Float16); // 32.8 MB
    const size_t mf_off    = xtB_off + xtB_sz;

    _Float16* packA = (_Float16*)(ws + packA_off);
    _Float16* packB = (_Float16*)(ws + packB_off);
    _Float16* xtA   = (_Float16*)(ws + xtA_off);
    _Float16* xtB   = (_Float16*)(ws + xtB_off);
    _Float16* mf    = (_Float16*)(ws + mf_off);   // 16.4 MB

    // 1) pack weights + transform activations to channel-innermost f16
    pack_wgts<<<dim3((9 * 8 * 4 * 32 + 255) / 256), 256, 0, stream>>>(hm_w1, packA, CIN_A);
    pack_wgts<<<dim3((9 * 4 * 4 * 32 + 255) / 256), 256, 0, stream>>>(mask_w, packB, CIN_B);
    nhwc_f16<CIN_A, HW_A><<<dim3((NB * (HW_A + 1) + 255) / 256), 256, 0, stream>>>(f_hm, xtA);
    nhwc_f16<CIN_B, HW_B><<<dim3((NB * (HW_B + 1) + 255) / 256), 256, 0, stream>>>(f_mask, xtB);
    // 2) fused heatmap branch (conv1+ReLU+conv2)
    conv_hm_fused<<<dim3(NB * 4), 256, 0, stream>>>(xtA, packA, hm_b1, hm_w2, hm_b2, hm_out);
    // 3) mask-feature conv -> ws (f16)
    conv_mask<<<dim3(NB * 16), 256, 0, stream>>>(xtB, packB, mask_b, mf);
    // 4) dynamic heads
    dyn_heads<<<dim3((HW_B + 255) / 256, NINST), 256, 0, stream>>>(mf, dynp, inst, masks, regs);
}
